// TransformerSpeaker_70961449664834
// MI455X (gfx1250) — matrix operand order repaired and checked by reading, not compiled, not run on hardware
//
#include <hip/hip_runtime.h>
#include <hip/hip_bf16.h>

// ---------------- problem constants ----------------
constexpr int D_    = 768;
constexpr int V_    = 2048;
constexpr int H_    = 12;
constexpr int NL_   = 6;
constexpr int FIMG_ = 49;
constexpr int B_    = 256;
constexpr int ML_   = 16;
constexpr int DFF_  = 3072;

// output region offsets (floats) in d_out
constexpr size_t OH0 = 0;                                    // (17,B,V)
constexpr size_t SEQO = OH0 + (size_t)(ML_ + 1) * B_ * V_;   // (B,17)
constexpr size_t LOGO = SEQO + (size_t)B_ * (ML_ + 1);       // (B,17)
constexpr size_t ENTO = LOGO + (size_t)B_ * (ML_ + 1);       // (B,17)
constexpr size_t OHT  = ENTO + (size_t)B_ * (ML_ + 1);       // (B,17,V)

typedef __attribute__((ext_vector_type(16))) __bf16 v16bf;
typedef __attribute__((ext_vector_type(2)))  __bf16 v2bf;
typedef __attribute__((ext_vector_type(8)))  float  v8f;

// pack two f32 -> one dword of two bf16 (single v_cvt_pk_bf16_f32)
__device__ inline unsigned int pack_bf16(float lo, float hi) {
  v2bf p;
  p.x = (__bf16)lo;
  p.y = (__bf16)hi;
  return __builtin_bit_cast(unsigned int, p);
}

// CDNA5 async global->LDS copy, 32 bytes per lane (two b128 ops).
// LDS dest address = low 32 bits of the flat shared pointer (ISA 10.2).
__device__ inline void async_b32bytes_to_lds(void* lds_ptr, const void* gaddr) {
  unsigned ldsb = (unsigned)(size_t)lds_ptr;
  asm volatile("global_load_async_to_lds_b128 %0, %1, off"
               :: "v"(ldsb), "v"(gaddr) : "memory");
  asm volatile("global_load_async_to_lds_b128 %0, %1, off offset:16"
               :: "v"(ldsb), "v"(gaddr) : "memory");
}
__device__ inline void wait_async_lds() {
  asm volatile("s_wait_asynccnt 0x0" ::: "memory");
}

// ---------------- wave / block reductions (wave32) ----------------
__device__ inline float wave_sum(float x) {
  #pragma unroll
  for (int o = 16; o > 0; o >>= 1) x += __shfl_xor(x, o, 32);
  return x;
}
__device__ inline float wave_max(float x) {
  #pragma unroll
  for (int o = 16; o > 0; o >>= 1) x = fmaxf(x, __shfl_xor(x, o, 32));
  return x;
}
__device__ inline float block_sum(float x, float* s) {  // blockDim.x == 256
  x = wave_sum(x);
  if ((threadIdx.x & 31) == 0) s[threadIdx.x >> 5] = x;
  __syncthreads();
  float r = 0.f;
  #pragma unroll
  for (int i = 0; i < 8; i++) r += s[i];
  __syncthreads();
  return r;
}
__device__ inline float block_max(float x, float* s) {
  x = wave_max(x);
  if ((threadIdx.x & 31) == 0) s[threadIdx.x >> 5] = x;
  __syncthreads();
  float r = s[0];
  #pragma unroll
  for (int i = 1; i < 8; i++) r = fmaxf(r, s[i]);
  __syncthreads();
  return r;
}
__device__ inline void block_argmax(float& val, int& idx, float* sv, int* si) {
  #pragma unroll
  for (int o = 16; o > 0; o >>= 1) {
    float ov = __shfl_xor(val, o, 32);
    int   oi = __shfl_xor(idx, o, 32);
    if (ov > val || (ov == val && oi < idx)) { val = ov; idx = oi; }
  }
  if ((threadIdx.x & 31) == 0) { sv[threadIdx.x >> 5] = val; si[threadIdx.x >> 5] = idx; }
  __syncthreads();
  val = sv[0]; idx = si[0];
  #pragma unroll
  for (int i = 1; i < 8; i++)
    if (sv[i] > val || (sv[i] == val && si[i] < idx)) { val = sv[i]; idx = si[i]; }
  __syncthreads();
}

// ---------------- fp32 -> bf16 weight conversion ----------------
__global__ void cvt_bf16_kernel(__bf16* __restrict__ dst, const float* __restrict__ src, int n) {
  int i = blockIdx.x * 256 + threadIdx.x;
  if (i < n) dst[i] = (__bf16)src[i];
}

// ---------------- WMMA bf16 GEMM: C[M,N] = A[M,K] @ W[N,K]^T + bias ----------------
// FLAGS bit0: relu, bit1: store bf16. N,K compile-time.
// block tile 128x128, BK=32, 256 threads = 8 waves, wave tile 32x64 (2x4 of 16x16).
// Double-buffered LDS. A: register prefetch + cvt to bf16. B: CDNA5 async
// global->LDS DMA (no VGPR round-trip, tracked by ASYNCcnt).
#define LDK 40   // 32 + 8 bf16 padding to dodge LDS bank conflicts
template <int FLAGS, int N, int K>
__global__ __launch_bounds__(256, 1)
void gemm_bf16_kernel(const float* __restrict__ A, const __bf16* __restrict__ Wb,
                      const float* __restrict__ bias, float* __restrict__ C,
                      __bf16* __restrict__ Cb) {
  __shared__ __bf16 As[2][128 * LDK];
  __shared__ __bf16 Bs[2][128 * LDK];

  const int tid  = threadIdx.x;
  const int lane = tid & 31;
  const int wid  = tid >> 5;
  const int wm   = wid & 3;   // 4 waves along M, 32 rows each
  const int wn   = wid >> 2;  // 2 waves along N, 64 cols each
  const int rowBase = blockIdx.y * 128;
  const int colBase = blockIdx.x * 128;

  v8f acc[2][4];
  #pragma unroll
  for (int m = 0; m < 2; m++)
    #pragma unroll
    for (int n = 0; n < 4; n++)
      #pragma unroll
      for (int j = 0; j < 8; j++) acc[m][n][j] = 0.f;

  // cooperative tile fill mapping: 2 threads per row, 16 elems each
  const int frow = tid >> 1;
  const int fk   = (tid & 1) * 16;
  const float*  Arow = A  + (size_t)(rowBase + frow) * K + fk;
  const __bf16* Brow = Wb + (size_t)(colBase + frow) * K + fk;

  uint4 apre[2];   // 16 bf16 of A, already converted
  auto gloadA = [&](int k0) {
    const float4* a4 = (const float4*)(Arow + k0);
    #pragma unroll
    for (int j = 0; j < 2; j++) {
      float4 t0 = a4[j * 2 + 0];
      float4 t1 = a4[j * 2 + 1];
      apre[j].x = pack_bf16(t0.x, t0.y);
      apre[j].y = pack_bf16(t0.z, t0.w);
      apre[j].z = pack_bf16(t1.x, t1.y);
      apre[j].w = pack_bf16(t1.z, t1.w);
    }
  };
  auto sstoreA = [&](int buf) {
    uint4* AsP = (uint4*)&As[buf][frow * LDK + fk];
    AsP[0] = apre[0];
    AsP[1] = apre[1];
  };
  auto asyncB = [&](int k0, int buf) {
    async_b32bytes_to_lds(&Bs[buf][frow * LDK + fk], Brow + k0);
  };

  // A fragment layout (16-bit A 16x32): lanes 0-15 -> K {0..7,16..23}; lanes 16-31 -> K {8..15,24..31}
  const int r  = lane & 15;
  const int ka = (lane < 16) ? 0 : 8;
  // B fragment layout (16-bit B 32x16): lanes 0-15 -> K {0..7,16..23}; lanes 16-31 -> K {8..15,24..31}
  const int kb = (lane < 16) ? 0 : 8;

  auto compute = [&](int buf) {
    const __bf16* AsB = As[buf];
    const __bf16* BsB = Bs[buf];
    union Frag { v16bf v; uint4 q[2]; };
    Frag af[2];
    #pragma unroll
    for (int m = 0; m < 2; m++) {
      const __bf16* p = &AsB[(wm * 32 + m * 16 + r) * LDK];
      af[m].q[0] = *(const uint4*)(p + ka);
      af[m].q[1] = *(const uint4*)(p + ka + 16);
    }
    #pragma unroll
    for (int n = 0; n < 4; n++) {
      Frag bfr;  // one B fragment live at a time (8 VGPRs)
      const __bf16* p = &BsB[(wn * 64 + n * 16 + r) * LDK];
      bfr.q[0] = *(const uint4*)(p + kb);
      bfr.q[1] = *(const uint4*)(p + kb + 16);
      #pragma unroll
      for (int m = 0; m < 2; m++)
        acc[m][n] = __builtin_amdgcn_wmma_f32_16x16x32_bf16(
            false, af[m].v, false, bfr.v, (short)0, acc[m][n], false, false);
    }
  };

  // prologue: fill buffer 0
  asyncB(0, 0);
  gloadA(0);
  sstoreA(0);
  wait_async_lds();
  __syncthreads();

  int buf = 0;
  #pragma unroll 1
  for (int k0 = 0; k0 < K - 32; k0 += 32) {  // steady state: always a next tile
    asyncB(k0 + 32, buf ^ 1);  // DMA B straight into the other LDS buffer
    gloadA(k0 + 32);           // A prefetch into regs (needs cvt)
    compute(buf);
    sstoreA(buf ^ 1);
    wait_async_lds();          // B DMA for next buffer complete
    __syncthreads();
    buf ^= 1;
  }
  compute(buf);                // last tile, no prefetch

  // epilogue: C/D 16x16 layout: VGPR j, lanes0-15 -> M=j, lanes16-31 -> M=j+8; N = lane&15
  const int mhalf = (lane >> 4) << 3;
  const int colW  = colBase + wn * 64 + (lane & 15);
  const int rowW  = rowBase + wm * 32 + mhalf;
  #pragma unroll
  for (int m = 0; m < 2; m++) {
    #pragma unroll
    for (int n = 0; n < 4; n++) {
      const int col = colW + n * 16;
      const float bv = bias[col];
      const int row0 = rowW + m * 16;
      if constexpr (FLAGS & 2) {
        __bf16* cp = Cb + (size_t)row0 * N + col;
        #pragma unroll
        for (int j = 0; j < 8; j++) cp[j * N] = (__bf16)(acc[m][n][j] + bv);
      } else {
        float* cp = C + (size_t)row0 * N + col;
        #pragma unroll
        for (int j = 0; j < 8; j++) {
          float val = acc[m][n][j] + bv;
          if constexpr (FLAGS & 1) val = fmaxf(val, 0.f);
          cp[j * N] = val;
        }
      }
    }
  }
}

// ---------------- softmax attention: one wave per (q,b,h), dh=64 ----------------
template <typename KT>
__global__ __launch_bounds__(256)
void attn_kernel(const float* __restrict__ q, const KT* __restrict__ k,
                 const KT* __restrict__ v, float* __restrict__ o, int Lq, int Lk) {
  const int gw   = blockIdx.x * 8 + (threadIdx.x >> 5);
  const int lane = threadIdx.x & 31;
  if (gw >= Lq * B_ * H_) return;
  const int h  = gw % H_;
  const int b  = (gw / H_) % B_;
  const int iq = gw / (H_ * B_);

  const float* qp = q + ((size_t)iq * B_ + b) * D_ + h * 64;
  const float q0 = qp[lane], q1 = qp[lane + 32];
  const float scale = 0.125f;  // 1/sqrt(64)

  float s[2] = {-1e30f, -1e30f};
  float smax = -1e30f;
  for (int kk = 0; kk < Lk; kk++) {
    const KT* kp = k + ((size_t)kk * B_ + b) * D_ + h * 64;
    float d = q0 * (float)kp[lane] + q1 * (float)kp[lane + 32];
    d = wave_sum(d) * scale;
    smax = fmaxf(smax, d);
    if ((kk & 31) == lane) s[kk >> 5] = d;
  }
  const float e0 = (lane < Lk)      ? __expf(s[0] - smax) : 0.f;
  const float e1 = (lane + 32 < Lk) ? __expf(s[1] - smax) : 0.f;
  const float inv = 1.f / wave_sum(e0 + e1);

  float o0 = 0.f, o1 = 0.f;
  for (int kk = 0; kk < Lk; kk++) {
    const float w = __shfl((kk < 32) ? e0 : e1, kk & 31, 32) * inv;
    const KT* vp = v + ((size_t)kk * B_ + b) * D_ + h * 64;
    o0 += w * (float)vp[lane];
    o1 += w * (float)vp[lane + 32];
  }
  float* op = o + ((size_t)iq * B_ + b) * D_ + h * 64;
  op[lane] = o0; op[lane + 32] = o1;
}

// ---------------- x = LayerNorm(x + res) * g + b, one block per row ----------------
__global__ __launch_bounds__(256)
void ln_res_kernel(float* __restrict__ x, const float* __restrict__ res,
                   const float* __restrict__ g, const float* __restrict__ bt, int rows) {
  __shared__ float sred[8];
  const int row = blockIdx.x;
  if (row >= rows) return;
  float*       xp = x   + (size_t)row * D_;
  const float* rp = res + (size_t)row * D_;
  float v[3]; float sum = 0.f;
  #pragma unroll
  for (int j = 0; j < 3; j++) { int d = threadIdx.x + j * 256; v[j] = xp[d] + rp[d]; sum += v[j]; }
  const float mean = block_sum(sum, sred) * (1.f / D_);
  float var = 0.f;
  #pragma unroll
  for (int j = 0; j < 3; j++) { float t = v[j] - mean; var += t * t; }
  var = block_sum(var, sred) * (1.f / D_);
  const float inv = rsqrtf(var + 1e-5f);
  #pragma unroll
  for (int j = 0; j < 3; j++) {
    int d = threadIdx.x + j * 256;
    xp[d] = (v[j] - mean) * inv * g[d] + bt[d];
  }
}

// ---------------- out[row] = in[row] + pos[row / B] ----------------
__global__ __launch_bounds__(256)
void add_pos_kernel(float* __restrict__ o, const float* __restrict__ in,
                    const float* __restrict__ pos, int rows) {
  const int row = blockIdx.x;
  if (row >= rows) return;
  const float* pp = pos + (size_t)(row / B_) * D_;
  #pragma unroll
  for (int j = 0; j < 3; j++) {
    int d = threadIdx.x + j * 256;
    o[(size_t)row * D_ + d] = in[(size_t)row * D_ + d] + pp[d];
  }
}

__global__ __launch_bounds__(256)
void bcast_bos_kernel(float* __restrict__ o, const float* __restrict__ bos) {
  const int b = blockIdx.x;
  #pragma unroll
  for (int j = 0; j < 3; j++) {
    int d = threadIdx.x + j * 256;
    o[(size_t)b * D_ + d] = bos[d];
  }
}

// ---------------- fused log_softmax + entropy + gumbel-softmax + sampling ----------------
__global__ __launch_bounds__(256)
void sample_kernel(const float* __restrict__ hl, const float* __restrict__ gum,
                   float* __restrict__ out, int step) {
  __shared__ float sred[8];
  __shared__ float sval[8];
  __shared__ int   sidx[8];
  const int b = blockIdx.x;
  const float* hp = hl  + (size_t)b * V_;
  const float* gp = gum + ((size_t)step * B_ + b) * V_;

  float h[8]; float m1 = -1e30f;
  #pragma unroll
  for (int j = 0; j < 8; j++) { h[j] = hp[threadIdx.x + j * 256]; m1 = fmaxf(m1, h[j]); }
  m1 = block_max(m1, sred);
  float se = 0.f;
  #pragma unroll
  for (int j = 0; j < 8; j++) se += __expf(h[j] - m1);
  se = block_sum(se, sred);
  const float lse = m1 + __logf(se);

  float ent = 0.f, m2 = -1e30f; float y[8];
  #pragma unroll
  for (int j = 0; j < 8; j++) {
    const float lp = h[j] - lse;
    ent -= __expf(lp) * lp;
    y[j] = lp + gp[threadIdx.x + j * 256];
    m2 = fmaxf(m2, y[j]);
  }
  ent = block_sum(ent, sred);
  m2  = block_max(m2, sred);
  float s2 = 0.f;
  #pragma unroll
  for (int j = 0; j < 8; j++) s2 += __expf(y[j] - m2);
  s2 = block_sum(s2, sred);
  const float inv2 = 1.f / s2;

  float bv = -1e30f; int bi = 0;
  #pragma unroll
  for (int j = 0; j < 8; j++) {
    const int vi = threadIdx.x + j * 256;
    if (y[j] > bv) { bv = y[j]; bi = vi; }
  }
  block_argmax(bv, bi, sval, sidx);

  float lpsum = 0.f;
  #pragma unroll
  for (int j = 0; j < 8; j++) {
    const int vi = threadIdx.x + j * 256;
    const float oh = __expf(y[j] - m2) * inv2;
    lpsum += __expf(h[j] - lse) * oh;
    out[OH0 + ((size_t)step * B_ + b) * V_ + vi] = oh;
    out[OHT + ((size_t)b * (ML_ + 1) + step) * V_ + vi] = oh;
  }
  lpsum = block_sum(lpsum, sred);
  if (threadIdx.x == 0) {
    out[SEQO + (size_t)b * (ML_ + 1) + step] = (float)bi;
    out[LOGO + (size_t)b * (ML_ + 1) + step] = __logf(lpsum + 1e-10f);
    out[ENTO + (size_t)b * (ML_ + 1) + step] = ent;
  }
}

// ---------------- pad rows (step 16) of all outputs ----------------
__global__ __launch_bounds__(256)
void init_pads_kernel(float* __restrict__ out) {
  const int i = blockIdx.x * 256 + threadIdx.x;
  if (i >= B_ * V_) return;
  const int b = i / V_, vv = i % V_;
  const float val = (vv == 0) ? 1.f : 0.f;
  out[OH0 + ((size_t)ML_ * B_ + b) * V_ + vv] = val;
  out[OHT + ((size_t)b * (ML_ + 1) + ML_) * V_ + vv] = val;
  if (vv == 0) {
    out[SEQO + (size_t)b * (ML_ + 1) + ML_] = 0.f;
    out[LOGO + (size_t)b * (ML_ + 1) + ML_] = 0.f;
    out[ENTO + (size_t)b * (ML_ + 1) + ML_] = 0.f;
  }
}

// ---------------- host orchestration ----------------
template <int FLAGS, int N, int K>
static inline void run_gemm(hipStream_t s, const float* A, const __bf16* W,
                            const float* bias, float* C, __bf16* Cb, int M) {
  dim3 g((unsigned)(N / 128), (unsigned)(M / 128)), blk(256);
  gemm_bf16_kernel<FLAGS, N, K><<<g, blk, 0, s>>>(A, W, bias, C, Cb);
}

// input index map (setup_inputs dict order, params flattened)
enum {
  IN_IMG = 0, IN_GUM, IN_IMG_POS, IN_TOK_POS, IN_BOS,
  IN_EMB_W, IN_EMB_B, IN_CLS_W, IN_CLS_B,
  IN_SA_WQ, IN_SA_WK, IN_SA_WV, IN_SA_WO,
  IN_SA_BQ, IN_SA_BK, IN_SA_BV, IN_SA_BO,
  IN_CA_WQ, IN_CA_WK, IN_CA_WV, IN_CA_WO,
  IN_CA_BQ, IN_CA_BK, IN_CA_BV, IN_CA_BO,
  IN_W1, IN_B1, IN_W2, IN_B2,
  IN_LN1G, IN_LN1B, IN_LN2G, IN_LN2B, IN_LN3G, IN_LN3B
};

extern "C" void kernel_launch(void* const* d_in, const int* in_sizes, int n_in,
                              void* d_out, int out_size, void* d_ws, size_t ws_size,
                              hipStream_t stream) {
  (void)in_sizes; (void)n_in; (void)out_size; (void)ws_size;
  auto F = [&](int i) { return (const float*)d_in[i]; };
  float* out = (float*)d_out;

  // workspace carving
  char* wsb = (char*)d_ws;
  size_t cur = 0;
  auto alloc = [&](size_t bytes) -> void* {
    void* p = wsb + cur;
    cur = (cur + bytes + 255) & ~(size_t)255;
    return p;
  };
  const size_t nAttnW = (size_t)NL_ * D_ * D_;
  const size_t nFfnW  = (size_t)NL_ * DFF_ * D_;
  const size_t nClsW  = (size_t)V_ * D_;
  const size_t nMemKV = (size_t)NL_ * FIMG_ * B_ * D_;

  __bf16* wb_saq = (__bf16*)alloc(nAttnW * 2);
  __bf16* wb_sak = (__bf16*)alloc(nAttnW * 2);
  __bf16* wb_sav = (__bf16*)alloc(nAttnW * 2);
  __bf16* wb_sao = (__bf16*)alloc(nAttnW * 2);
  __bf16* wb_caq = (__bf16*)alloc(nAttnW * 2);
  __bf16* wb_cak = (__bf16*)alloc(nAttnW * 2);
  __bf16* wb_cav = (__bf16*)alloc(nAttnW * 2);
  __bf16* wb_cao = (__bf16*)alloc(nAttnW * 2);
  __bf16* wb_w1  = (__bf16*)alloc(nFfnW * 2);
  __bf16* wb_w2  = (__bf16*)alloc(nFfnW * 2);
  __bf16* wb_cls = (__bf16*)alloc(nClsW * 2);
  __bf16* wb_emb = (__bf16*)alloc(nClsW * 2);
  __bf16* kmem   = (__bf16*)alloc(nMemKV * 2);
  __bf16* vmem   = (__bf16*)alloc(nMemKV * 2);
  float* mem  = (float*)alloc((size_t)FIMG_ * B_ * D_ * 4);
  float* xbuf = (float*)alloc((size_t)ML_ * B_ * D_ * 4);
  float* qbuf = (float*)alloc((size_t)ML_ * B_ * D_ * 4);
  float* kbuf = (float*)alloc((size_t)ML_ * B_ * D_ * 4);
  float* vbuf = (float*)alloc((size_t)ML_ * B_ * D_ * 4);
  float* obuf = (float*)alloc((size_t)ML_ * B_ * D_ * 4);
  float* tbuf = (float*)alloc((size_t)ML_ * B_ * D_ * 4);
  float* h1   = (float*)alloc((size_t)ML_ * B_ * DFF_ * 4);
  float* seqh = (float*)alloc((size_t)ML_ * B_ * D_ * 4);
  float* logb = (float*)alloc((size_t)B_ * V_ * 4);

  // pad rows of outputs
  init_pads_kernel<<<(B_ * V_ + 255) / 256, 256, 0, stream>>>(out);

  // convert weights to bf16 (once per call; bf16 weights ~120MB -> L2 resident)
  auto cvt = [&](__bf16* dst, const float* src, size_t n) {
    cvt_bf16_kernel<<<(unsigned)((n + 255) / 256), 256, 0, stream>>>(dst, src, (int)n);
  };
  cvt(wb_saq, F(IN_SA_WQ), nAttnW); cvt(wb_sak, F(IN_SA_WK), nAttnW);
  cvt(wb_sav, F(IN_SA_WV), nAttnW); cvt(wb_sao, F(IN_SA_WO), nAttnW);
  cvt(wb_caq, F(IN_CA_WQ), nAttnW); cvt(wb_cak, F(IN_CA_WK), nAttnW);
  cvt(wb_cav, F(IN_CA_WV), nAttnW); cvt(wb_cao, F(IN_CA_WO), nAttnW);
  cvt(wb_w1,  F(IN_W1),    nFfnW);  cvt(wb_w2,  F(IN_W2),    nFfnW);
  cvt(wb_cls, F(IN_CLS_W), nClsW);  cvt(wb_emb, F(IN_EMB_W), nClsW);

  // mem = image_features + img_pos
  add_pos_kernel<<<FIMG_ * B_, 256, 0, stream>>>(mem, F(IN_IMG), F(IN_IMG_POS), FIMG_ * B_);

  // precompute cross-attention K/V for all layers (step-invariant, saves ~40% of FLOPs)
  for (int l = 0; l < NL_; l++) {
    run_gemm<2, D_, D_>(stream, mem, wb_cak + (size_t)l * D_ * D_, F(IN_CA_BK) + (size_t)l * D_,
                        nullptr, kmem + (size_t)l * FIMG_ * B_ * D_, FIMG_ * B_);
    run_gemm<2, D_, D_>(stream, mem, wb_cav + (size_t)l * D_ * D_, F(IN_CA_BV) + (size_t)l * D_,
                        nullptr, vmem + (size_t)l * FIMG_ * B_ * D_, FIMG_ * B_);
  }

  // prev = bos
  bcast_bos_kernel<<<B_, 256, 0, stream>>>(seqh, F(IN_BOS));

  for (int i = 0; i < ML_; i++) {
    const int L = i + 1, Ntok = L * B_;
    add_pos_kernel<<<Ntok, 256, 0, stream>>>(xbuf, seqh, F(IN_TOK_POS), Ntok);

    for (int l = 0; l < NL_; l++) {
      const size_t wo  = (size_t)l * D_ * D_;
      const size_t wfo = (size_t)l * DFF_ * D_;
      // self-attention
      run_gemm<0, D_, D_>(stream, xbuf, wb_saq + wo, F(IN_SA_BQ) + (size_t)l * D_, qbuf, nullptr, Ntok);
      run_gemm<0, D_, D_>(stream, xbuf, wb_sak + wo, F(IN_SA_BK) + (size_t)l * D_, kbuf, nullptr, Ntok);
      run_gemm<0, D_, D_>(stream, xbuf, wb_sav + wo, F(IN_SA_BV) + (size_t)l * D_, vbuf, nullptr, Ntok);
      {
        int tw = L * B_ * H_;
        attn_kernel<float><<<(tw + 7) / 8, 256, 0, stream>>>(qbuf, kbuf, vbuf, obuf, L, L);
      }
      run_gemm<0, D_, D_>(stream, obuf, wb_sao + wo, F(IN_SA_BO) + (size_t)l * D_, tbuf, nullptr, Ntok);
      ln_res_kernel<<<Ntok, 256, 0, stream>>>(xbuf, tbuf, F(IN_LN1G) + (size_t)l * D_, F(IN_LN1B) + (size_t)l * D_, Ntok);
      // cross-attention (precomputed bf16 K/V)
      run_gemm<0, D_, D_>(stream, xbuf, wb_caq + wo, F(IN_CA_BQ) + (size_t)l * D_, qbuf, nullptr, Ntok);
      {
        int tw = L * B_ * H_;
        attn_kernel<__bf16><<<(tw + 7) / 8, 256, 0, stream>>>(
            qbuf, kmem + (size_t)l * FIMG_ * B_ * D_, vmem + (size_t)l * FIMG_ * B_ * D_, obuf, L, FIMG_);
      }
      run_gemm<0, D_, D_>(stream, obuf, wb_cao + wo, F(IN_CA_BO) + (size_t)l * D_, tbuf, nullptr, Ntok);
      ln_res_kernel<<<Ntok, 256, 0, stream>>>(xbuf, tbuf, F(IN_LN2G) + (size_t)l * D_, F(IN_LN2B) + (size_t)l * D_, Ntok);
      // FFN
      run_gemm<1, DFF_, D_>(stream, xbuf, wb_w1 + wfo, F(IN_B1) + (size_t)l * DFF_, h1, nullptr, Ntok);
      run_gemm<0, D_, DFF_>(stream, h1, wb_w2 + wfo, F(IN_B2) + (size_t)l * D_, tbuf, nullptr, Ntok);
      ln_res_kernel<<<Ntok, 256, 0, stream>>>(xbuf, tbuf, F(IN_LN3G) + (size_t)l * D_, F(IN_LN3B) + (size_t)l * D_, Ntok);
    }

    // classifier on the last position (rows i*B .. i*B+255, contiguous)
    run_gemm<0, V_, D_>(stream, xbuf + (size_t)i * B_ * D_, wb_cls, F(IN_CLS_B), logb, nullptr, B_);
    sample_kernel<<<B_, 256, 0, stream>>>(logb, F(IN_GUM), out, i);

    // prev = oh @ embed_W^T + embed_b  -> seq_hidden[i+1]  (skip unused final embed)
    if (i + 1 < ML_)
      run_gemm<0, D_, V_>(stream, out + OH0 + (size_t)i * B_ * V_, wb_emb, F(IN_EMB_B),
                          seqh + (size_t)(i + 1) * B_ * D_, nullptr, B_);
  }
}
